// RACL_Layer_78572131713537
// MI455X (gfx1250) — matrix operand order repaired and checked by reading, not compiled, not run on hardware
//
#include <hip/hip_runtime.h>
#include <hip/hip_bf16.h>

// MI455X / gfx1250 implementation of the RACL layer.
// All heavy math runs on v_wmma_f32_16x16x32_f16 (f16 operands, f32 accum).
// Wave32 blocks; each wave produces a 32x64 C macro-tile (8 WMMA tiles):
// 2 A fragments x 4 B fragments, so every B fragment feeds 2 WMMAs and every
// A fragment feeds 4 WMMAs. EXEC is all-ones at every WMMA.

typedef __attribute__((ext_vector_type(16))) _Float16 v16h;
typedef __attribute__((ext_vector_type(8)))  _Float16 v8h;
typedef __attribute__((ext_vector_type(8)))  float    v8f;

#define NB 16      // batch
#define LLEN 512   // sequence length
#define DDIM 768   // feature dim
#define NHOP 3

// ---------------------------------------------------------------------------
// WMMA fragment helpers (CDNA5 16x16x32 f16 layouts, CDNA5 ISA)
// A (16x32, MxK): lanes 0-15 row M=lane, K={k0..k0+7, k0+16..k0+23};
//                 lanes 16-31 row M=lane-16, K={k0+8..k0+15, k0+24..k0+31}.
// B (32x16, KxN) from Bt stored [N][K] row-major: lane&15 = N,
//                 K order matches A (lanes 0-15 K={k0..k0+7, k0+16..k0+23},
//                 lanes 16-31 K={k0+8..k0+15, k0+24..k0+31}).
// C/D (16x16 f32): vgpr j -> (M = j + 8*(lane>=16), N = lane&15).
// ---------------------------------------------------------------------------
__device__ __forceinline__ v16h load_a_u(const _Float16* __restrict__ A,
                                         int row0, int k0, int lda) {
  int lane = threadIdx.x & 31;
  int r  = row0 + (lane & 15);
  int kk = k0 + ((lane & 16) ? 8 : 0);
  const _Float16* p = A + (long)r * lda + kk;
  union { v16h v; v8h h[2]; } u;
  u.h[0] = *(const v8h*)(p);
  u.h[1] = *(const v8h*)(p + 16);
  return u.v;
}

// checked variant: zero-fills out-of-range rows (conv zero padding)
__device__ __forceinline__ v16h load_a_c(const _Float16* __restrict__ A,
                                         int row0, int k0, int lda, int rlim) {
  int lane = threadIdx.x & 31;
  int r  = row0 + (lane & 15);
  int kk = k0 + ((lane & 16) ? 8 : 0);
  union { v16h v; v8h h[2]; } u;
  if ((unsigned)r < (unsigned)rlim) {
    const _Float16* p = A + (long)r * lda + kk;
    u.h[0] = *(const v8h*)(p);
    u.h[1] = *(const v8h*)(p + 16);
  } else {
    v8h z = {};
    u.h[0] = z; u.h[1] = z;
  }
  return u.v;
}

__device__ __forceinline__ v16h load_b_frag(const _Float16* __restrict__ Bt,
                                            int n0, int k0, int ldb) {
  int lane = threadIdx.x & 31;
  int n  = n0 + (lane & 15);
  int kk = k0 + ((lane & 16) ? 8 : 0);
  const _Float16* p = Bt + (long)n * ldb + kk;
  union { v16h v; v8h h[2]; } u;
  u.h[0] = *(const v8h*)(p);
  u.h[1] = *(const v8h*)(p + 16);
  return u.v;
}

__device__ __forceinline__ v8f wmma16(v16h a, v16h b, v8f c) {
  return __builtin_amdgcn_wmma_f32_16x16x32_f16(false, a, false, b, (short)0, c,
                                                false, false);
}

// ---------------------------------------------------------------------------
// Generic batched GEMM: C[b] = A[b] (MxK, row-major) * Bt[b]^T (Bt is NxK).
// One wave -> 32x64 macro-tile of C (8 WMMA tiles, 2 A frags x 4 B frags).
// Used for S = X X^T (A=Bt=X) and O = P V (A=P, Bt=V^T).
// ---------------------------------------------------------------------------
__global__ void __launch_bounds__(32)
gemm_rr(const _Float16* __restrict__ A, const _Float16* __restrict__ Bt,
        float* __restrict__ C, int K, int lda, int ldb, int ldc,
        long sA, long sB, long sC) {
  int nt = blockIdx.x * 64;
  int mt = blockIdx.y * 32;
  const _Float16* Ab = A  + blockIdx.z * sA;
  const _Float16* Bb = Bt + blockIdx.z * sB;
  float*          Cb = C  + blockIdx.z * sC;
  v8f acc[8] = {{}, {}, {}, {}, {}, {}, {}, {}};
  for (int k = 0; k < K; k += 32) {
    __builtin_prefetch(Ab + (long)mt * lda + k + 32, 0, 1);  // speculative
    v16h a0 = load_a_u(Ab, mt, k, lda);
    v16h a1 = load_a_u(Ab, mt + 16, k, lda);
#pragma unroll
    for (int i = 0; i < 4; ++i) {
      v16h b = load_b_frag(Bb, nt + 16 * i, k, ldb);
      acc[i]     = wmma16(a0, b, acc[i]);
      acc[i + 4] = wmma16(a1, b, acc[i + 4]);
    }
  }
  int lane = threadIdx.x & 31;
  int nn = lane & 15;
  int mb = mt + ((lane & 16) ? 8 : 0);
#pragma unroll
  for (int r = 0; r < 2; ++r)
#pragma unroll
    for (int i = 0; i < 4; ++i)
#pragma unroll
      for (int j = 0; j < 8; ++j)
        Cb[(long)(mb + 16 * r + j) * ldc + nt + 16 * i + nn] = acc[4 * r + i][j];
}

// ---------------------------------------------------------------------------
// conv1d(k=3, pad=1) as 3 shifted GEMMs, + bias + ReLU.
// X: [L][D] f16 per batch. W: 3 packed [Dout][Din] f16 matrices.
// Interior row-tiles take the unchecked fast path; only the first/last
// L-tile (mt==0 / mt==480) pays for zero-padding row checks.
// Writes f32 (for l2norm / output "cur") and f16 (next hop's input).
// ---------------------------------------------------------------------------
__global__ void __launch_bounds__(32)
conv_gemm(const _Float16* __restrict__ X, const _Float16* __restrict__ W,
          const float* __restrict__ bias, float* __restrict__ outF,
          _Float16* __restrict__ outH) {
  int nt = blockIdx.x * 64;
  int mt = blockIdx.y * 32;
  int b  = blockIdx.z;
  const _Float16* Xb = X + (long)b * LLEN * DDIM;
  v8f acc[8] = {{}, {}, {}, {}, {}, {}, {}, {}};
  bool interior = (mt != 0) && (mt + 32 != LLEN);
  if (interior) {
#pragma unroll
    for (int t = 0; t < 3; ++t) {
      const _Float16* Wt = W + (long)t * DDIM * DDIM;
      for (int k = 0; k < DDIM; k += 32) {
        v16h a0 = load_a_u(Xb, mt + t - 1, k, DDIM);
        v16h a1 = load_a_u(Xb, mt + 16 + t - 1, k, DDIM);
#pragma unroll
        for (int i = 0; i < 4; ++i) {
          v16h w = load_b_frag(Wt, nt + 16 * i, k, DDIM);
          acc[i]     = wmma16(a0, w, acc[i]);
          acc[i + 4] = wmma16(a1, w, acc[i + 4]);
        }
      }
    }
  } else {
#pragma unroll
    for (int t = 0; t < 3; ++t) {
      const _Float16* Wt = W + (long)t * DDIM * DDIM;
      for (int k = 0; k < DDIM; k += 32) {
        v16h a0 = load_a_c(Xb, mt + t - 1, k, DDIM, LLEN);       // zero-pad
        v16h a1 = load_a_c(Xb, mt + 16 + t - 1, k, DDIM, LLEN);  // zero-pad
#pragma unroll
        for (int i = 0; i < 4; ++i) {
          v16h w = load_b_frag(Wt, nt + 16 * i, k, DDIM);
          acc[i]     = wmma16(a0, w, acc[i]);
          acc[i + 4] = wmma16(a1, w, acc[i + 4]);
        }
      }
    }
  }
  int lane = threadIdx.x & 31;
  int nn = lane & 15;
  int mb = mt + ((lane & 16) ? 8 : 0);
#pragma unroll
  for (int r = 0; r < 2; ++r)
#pragma unroll
    for (int i = 0; i < 4; ++i) {
      int n = nt + 16 * i + nn;
      float bn = bias[n];
#pragma unroll
      for (int j = 0; j < 8; ++j) {
        float v = acc[4 * r + i][j] + bn;
        v = v > 0.f ? v : 0.f;                                   // ReLU
        long idx = ((long)b * LLEN + (mb + 16 * r + j)) * DDIM + n;
        outF[idx] = v;
        outH[idx] = (_Float16)v;
      }
    }
}

// ---------------------------------------------------------------------------
// Pooling GEMM: scores[m] += sum_n tanh((R W1)[m,n] + b1[n]) * w2[n]
// R: [32768 x 768] f16, W1t: [768 x 768] f16 (= W1^T). 32x64 macro-tile per
// wave; the n-dot is reduced across each 16-lane half-wave then atomically
// accumulated into the per-position score.
// ---------------------------------------------------------------------------
__global__ void __launch_bounds__(32)
pool_gemm(const _Float16* __restrict__ R, const _Float16* __restrict__ W1t,
          const float* __restrict__ b1, const float* __restrict__ w2,
          float* __restrict__ scores) {
  int nt = blockIdx.x * 64;
  int mt = blockIdx.y * 32;
  v8f acc[8] = {{}, {}, {}, {}, {}, {}, {}, {}};
  for (int k = 0; k < DDIM; k += 32) {
    v16h a0 = load_a_u(R, mt, k, DDIM);
    v16h a1 = load_a_u(R, mt + 16, k, DDIM);
#pragma unroll
    for (int i = 0; i < 4; ++i) {
      v16h b = load_b_frag(W1t, nt + 16 * i, k, DDIM);
      acc[i]     = wmma16(a0, b, acc[i]);
      acc[i + 4] = wmma16(a1, b, acc[i + 4]);
    }
  }
  int lane = threadIdx.x & 31;
  int nn = lane & 15;
  int mb = mt + ((lane & 16) ? 8 : 0);
#pragma unroll
  for (int r = 0; r < 2; ++r)
#pragma unroll
    for (int i = 0; i < 4; ++i) {
      int n = nt + 16 * i + nn;
      float b1n = b1[n];
      float w2n = w2[n];
#pragma unroll
      for (int j = 0; j < 8; ++j) {
        float v = tanhf(acc[4 * r + i][j] + b1n) * w2n;
#pragma unroll
        for (int s = 1; s < 16; s <<= 1)      // reduce within 16-lane half
          v += __shfl_xor(v, s, 32);
        if ((lane & 15) == 0) atomicAdd(&scores[mb + 16 * r + j], v);
      }
    }
}

// ---------------------------------------------------------------------------
// Elementwise / reduction helpers
// ---------------------------------------------------------------------------
__global__ void cvt_f32_f16(const float* __restrict__ in,
                            _Float16* __restrict__ out, long n) {
  long i = (long)blockIdx.x * blockDim.x + threadIdx.x;
  if (i < n) out[i] = (_Float16)in[i];
}

// out[b][c][r] = (f16) in[b][r][c]
__global__ void transpose_cvt(const float* __restrict__ in,
                              _Float16* __restrict__ out,
                              int rows, int cols, long n) {
  long i = (long)blockIdx.x * blockDim.x + threadIdx.x;
  if (i >= n) return;
  long rc  = (long)rows * cols;
  long b   = i / rc;
  long rem = i - b * rc;
  int  c   = (int)(rem / rows);
  int  r   = (int)(rem - (long)c * rows);
  out[i] = (_Float16)in[b * rc + (long)r * cols + c];
}

// Wpk[h][t][o][i] = (f16) conv_w[h][o][i][t]
__global__ void pack_convw(const float* __restrict__ w,
                           _Float16* __restrict__ out) {
  long i = (long)blockIdx.x * blockDim.x + threadIdx.x;
  const long n = 3L * 3 * DDIM * DDIM;
  if (i >= n) return;
  long hh   = i / (3L * DDIM * DDIM);
  long rem  = i - hh * (3L * DDIM * DDIM);
  long t    = rem / ((long)DDIM * DDIM);
  long rem2 = rem - t * ((long)DDIM * DDIM);
  long o    = rem2 / DDIM;
  long ii   = rem2 - o * DDIM;
  out[i] = (_Float16)w[((hh * DDIM + o) * DDIM + ii) * 3 + t];
}

// softmask per query row (one block = one (b,l), 256 thr, keys = 512)
__global__ void __launch_bounds__(256)
softmask_rows(const float* __restrict__ S, const float* __restrict__ mask,
              _Float16* __restrict__ P) {
  __shared__ float red[256];
  int bl = blockIdx.x;
  int b  = bl >> 9;
  int l  = bl & 511;
  const float* row = S + (long)bl * LLEN;
  const float* mk  = mask + (long)b * LLEN;
  int t = threadIdx.x;
  float x0 = row[t], x1 = row[t + 256];
  red[t] = fmaxf(x0, x1);
  __syncthreads();
  for (int s = 128; s > 0; s >>= 1) {
    if (t < s) red[t] = fmaxf(red[t], red[t + s]);
    __syncthreads();
  }
  float mx = red[0];
  __syncthreads();
  float y0 = __expf(x0 - mx) * mk[t];
  float y1 = __expf(x1 - mx) * mk[t + 256];
  red[t] = y0 + y1;
  __syncthreads();
  for (int s = 128; s > 0; s >>= 1) {
    if (t < s) red[t] += red[t + s];
    __syncthreads();
  }
  float inv = mk[l] / (red[0] + 1e-10f);
  P[(long)bl * LLEN + t]       = (_Float16)(y0 * inv);
  P[(long)bl * LLEN + t + 256] = (_Float16)(y1 * inv);
}

// l2 row normalize: N = x / max(||x||, 1e-12) (one block per (b,l))
__global__ void __launch_bounds__(256)
l2norm_rows(const float* __restrict__ X, _Float16* __restrict__ N) {
  __shared__ float red[256];
  int bl = blockIdx.x;
  const float* row = X + (long)bl * DDIM;
  int t = threadIdx.x;
  float a0 = row[t], a1 = row[t + 256], a2 = row[t + 512];
  red[t] = a0 * a0 + a1 * a1 + a2 * a2;
  __syncthreads();
  for (int s = 128; s > 0; s >>= 1) {
    if (t < s) red[t] += red[t + s];
    __syncthreads();
  }
  float inv = 1.f / fmaxf(sqrtf(red[0]), 1e-12f);
  N[(long)bl * DDIM + t]       = (_Float16)(a0 * inv);
  N[(long)bl * DDIM + t + 256] = (_Float16)(a1 * inv);
  N[(long)bl * DDIM + t + 512] = (_Float16)(a2 * inv);
}

__global__ void init_scores(float* __restrict__ s, const float* __restrict__ b2,
                            int n) {
  int i = blockIdx.x * blockDim.x + threadIdx.x;
  if (i < n) s[i] = b2[0];
}

// softmax over 4 hop scores, weighted sum of result slots; emit (cur, weighted)
__global__ void finalize(const float* __restrict__ res,
                         const float* __restrict__ sc,
                         const float* __restrict__ cur,
                         float* __restrict__ out, long n) {
  long idx = (long)blockIdx.x * blockDim.x + threadIdx.x;
  if (idx >= n) return;
  long bl = idx / DDIM;
  int  d  = (int)(idx - bl * DDIM);
  const float* s = sc + bl * 4;
  float s0 = s[0], s1 = s[1], s2 = s[2], s3 = s[3];
  float m  = fmaxf(fmaxf(s0, s1), fmaxf(s2, s3));
  float e0 = __expf(s0 - m), e1 = __expf(s1 - m);
  float e2 = __expf(s2 - m), e3 = __expf(s3 - m);
  float inv = 1.f / (e0 + e1 + e2 + e3);
  const float* r = res + bl * 4 * DDIM + d;
  float w = (e0 * r[0] + e1 * r[DDIM] + e2 * r[2 * DDIM] + e3 * r[3 * DDIM]) * inv;
  out[idx]     = cur[idx];   // first output: cur (last hop's conv)
  out[n + idx] = w;          // second output: weighted
}

// ---------------------------------------------------------------------------
extern "C" void kernel_launch(void* const* d_in, const int* in_sizes, int n_in,
                              void* d_out, int out_size, void* d_ws,
                              size_t ws_size, hipStream_t stream) {
  const float* inputs  = (const float*)d_in[0];
  const float* mask    = (const float*)d_in[1];
  const float* conv_w  = (const float*)d_in[2];
  const float* conv_b  = (const float*)d_in[3];
  const float* pool_w1 = (const float*)d_in[4];
  const float* pool_b1 = (const float*)d_in[5];
  const float* pool_w2 = (const float*)d_in[6];
  const float* pool_b2 = (const float*)d_in[7];
  // d_in[8] = position (unused by the module)

  const long BLD = (long)NB * LLEN * DDIM;          // 6,291,456
  char* ws = (char*)d_ws;
  // workspace layout (all offsets 256B aligned); total ~204 MB
  float*    res    = (float*)   (ws + 0L);           // [B][L][4][D] f32
  float*    conv   = (float*)   (ws + 100663296L);   // [B][L][D]    f32
  float*    S      = (float*)   (ws + 125829120L);   // [B][L][L]    f32
  _Float16* Ph     = (_Float16*)(ws + 142606336L);   // [B][L][L]    f16
  _Float16* Wpk    = (_Float16*)(ws + 150994944L);   // [3][3][D][D] f16
  _Float16* W1t    = (_Float16*)(ws + 161611776L);   // [D][D]       f16
  float*    pscore = (float*)   (ws + 162791424L);   // [B*L*4]      f32
  _Float16* hbase  = (_Float16*)(ws + 163315712L);   // 4 x BLD f16 region
  _Float16* Xh  = hbase;                             // f16(cur)
  _Float16* Yh  = hbase + BLD;                       // f16(conv) -> next cur
  _Float16* Nh  = hbase + 2 * BLD;                   // f16(l2norm(conv))
  _Float16* Vth = hbase + 3 * BLD;                   // f16(V^T) [B][D][L]
  _Float16* Rh  = hbase;                             // alias after hop loop

  const long sX = (long)LLEN * DDIM;                 // per-batch operand stride
  const long sS = (long)LLEN * LLEN;
  const long sR = (long)LLEN * 4 * DDIM;

  // --- operand prep -------------------------------------------------------
  cvt_f32_f16<<<(BLD + 255) / 256, 256, 0, stream>>>(inputs, Xh, BLD);
  transpose_cvt<<<(BLD + 255) / 256, 256, 0, stream>>>(inputs, Vth, LLEN, DDIM, BLD);
  pack_convw<<<(3L * 3 * DDIM * DDIM + 255) / 256, 256, 0, stream>>>(conv_w, Wpk);
  transpose_cvt<<<((long)DDIM * DDIM + 255) / 256, 256, 0, stream>>>(
      pool_w1, W1t, DDIM, DDIM, (long)DDIM * DDIM);

  // --- hop 0: raw similarity attention ------------------------------------
  gemm_rr<<<dim3(LLEN / 64, LLEN / 32, NB), 32, 0, stream>>>(
      Xh, Xh, S, DDIM, DDIM, DDIM, LLEN, sX, sX, sS);
  softmask_rows<<<NB * LLEN, 256, 0, stream>>>(S, mask, Ph);
  gemm_rr<<<dim3(DDIM / 64, LLEN / 32, NB), 32, 0, stream>>>(
      Ph, Vth, res + 0 * DDIM, LLEN, LLEN, LLEN, 4 * DDIM, sS, sX, sR);

  // --- hops 1..3: conv -> l2norm -> cosine attention -----------------------
  _Float16* Xc = Xh;
  _Float16* Yc = Yh;
  for (int h = 0; h < NHOP; ++h) {
    conv_gemm<<<dim3(DDIM / 64, LLEN / 32, NB), 32, 0, stream>>>(
        Xc, Wpk + (long)h * 3 * DDIM * DDIM, conv_b + h * DDIM, conv, Yc);
    l2norm_rows<<<NB * LLEN, 256, 0, stream>>>(conv, Nh);
    transpose_cvt<<<(BLD + 255) / 256, 256, 0, stream>>>(conv, Vth, LLEN, DDIM, BLD);
    gemm_rr<<<dim3(LLEN / 64, LLEN / 32, NB), 32, 0, stream>>>(
        Nh, Nh, S, DDIM, DDIM, DDIM, LLEN, sX, sX, sS);
    softmask_rows<<<NB * LLEN, 256, 0, stream>>>(S, mask, Ph);
    gemm_rr<<<dim3(DDIM / 64, LLEN / 32, NB), 32, 0, stream>>>(
        Ph, Vth, res + (h + 1) * DDIM, LLEN, LLEN, LLEN, 4 * DDIM, sS, sX, sR);
    _Float16* tmp = Xc; Xc = Yc; Yc = tmp;           // conv becomes next cur
  }

  // --- pooling over the 4 hop outputs --------------------------------------
  cvt_f32_f16<<<(4 * BLD + 255) / 256, 256, 0, stream>>>(res, Rh, 4 * BLD);
  init_scores<<<(NB * LLEN * 4 + 255) / 256, 256, 0, stream>>>(
      pscore, pool_b2, NB * LLEN * 4);
  pool_gemm<<<dim3(DDIM / 64, (NB * LLEN * 4) / 32), 32, 0, stream>>>(
      Rh, W1t, pool_b1, pool_w2, pscore);
  finalize<<<(BLD + 255) / 256, 256, 0, stream>>>(
      res, pscore, conv, (float*)d_out, BLD);
}
